// GINE_Allocation_Predictor_82609400971330
// MI455X (gfx1250) — matrix operand order repaired and checked by reading, not compiled, not run on hardware
//
#include <hip/hip_runtime.h>
#include <math.h>

#define FDIM 9
#define HDIM 64

typedef __attribute__((ext_vector_type(16))) _Float16 v16h;
typedef __attribute__((ext_vector_type(8)))  _Float16 v8h;
typedef __attribute__((ext_vector_type(8)))  float    v8f;

__device__ __forceinline__ void atomAddF32(float* p, float v) {
  (void)__hip_atomic_fetch_add(p, v, __ATOMIC_RELAXED, __HIP_MEMORY_SCOPE_AGENT);
}

// ---- WMMA fragment loaders (CDNA5 wave32 16-bit layouts, ISA 7.12.2) ----
// A: 16x32 f16, row m0+r; lanes 0-15 get K=kc+[0..7],[16..23]; lanes 16-31 K=kc+[8..15],[24..31]
__device__ __forceinline__ v16h fragA(const _Float16* A, int lda, int m0, int kc, int r, int h) {
  const _Float16* row = A + (m0 + r) * lda;
  v8h lo = *(const v8h*)(row + kc + (h ? 8 : 0));
  v8h hi = *(const v8h*)(row + kc + (h ? 24 : 16));
  return __builtin_shufflevector(lo, hi, 0,1,2,3,4,5,6,7,8,9,10,11,12,13,14,15);
}
// B: 32x16 f16 from transposed weights WT[col][k]; lane holds col n0+r, K=kc+8h+[0..7],[16..23]
__device__ __forceinline__ v16h fragB(const _Float16* WT, int ldb, int n0, int kc, int r, int h) {
  const _Float16* col = WT + (n0 + r) * ldb + kc + (h ? 8 : 0);
  v8h lo = *(const v8h*)(col);
  v8h hi = *(const v8h*)(col + 16);
  return __builtin_shufflevector(lo, hi, 0,1,2,3,4,5,6,7,8,9,10,11,12,13,14,15);
}

__device__ __forceinline__ v8f gemm_tile(const _Float16* A, int lda, const _Float16* WT, int ldb,
                                         int m0, int n0, int K, int r, int h) {
  v8f acc = {};
  for (int kc = 0; kc < K; kc += 32) {
    v16h a = fragA(A, lda, m0, kc, r, h);
    v16h b = fragB(WT, ldb, n0, kc, r, h);
    acc = __builtin_amdgcn_wmma_f32_16x16x32_f16(false, a, false, b, (short)0, acc, false, false);
  }
  return acc;
}

// ---- Edge phase 1: agg1[dst] += relu(x[src] + attr*We1 + be1), F=9 ----
__global__ __launch_bounds__(256) void edge1_kernel(const float* __restrict__ x,
    const int* __restrict__ ei, const float* __restrict__ ea,
    const float* __restrict__ We1, const float* __restrict__ be1,
    float* __restrict__ agg1, int E) {
  int e = blockIdx.x * 256 + threadIdx.x;
  if (e >= E) return;
  int s = ei[e], d = ei[E + e];
  float a = ea[e];
  const float* xr = x + (size_t)s * FDIM;
  float* ag = agg1 + (size_t)d * FDIM;
#pragma unroll
  for (int f = 0; f < FDIM; ++f) {
    float m = fmaxf(fmaf(We1[f], a, xr[f] + be1[f]), 0.f);
    atomAddF32(&ag[f], m);
  }
}

// ---- Edge phase 2: 16 lanes per edge, float4 gather of h1 row + 4 atomics ----
__global__ __launch_bounds__(256) void edge2_kernel(const float* __restrict__ h1,
    const int* __restrict__ ei, const float* __restrict__ ea,
    const float* __restrict__ We2, const float* __restrict__ be2,
    float* __restrict__ agg2, int E) {
  int t = blockIdx.x * 256 + threadIdx.x;
  int e = t >> 4;
  if (e >= E) return;
  int fb = (t & 15) * 4;
  int s = ei[e], d = ei[E + e];
  float a = ea[e];
  float4 hv = *(const float4*)(h1 + (size_t)s * HDIM + fb);
  float4 wv = *(const float4*)(We2 + fb);
  float4 bv = *(const float4*)(be2 + fb);
  float* ag = agg2 + (size_t)d * HDIM + fb;
  atomAddF32(ag + 0, fmaxf(fmaf(wv.x, a, hv.x + bv.x), 0.f));
  atomAddF32(ag + 1, fmaxf(fmaf(wv.y, a, hv.y + bv.y), 0.f));
  atomAddF32(ag + 2, fmaxf(fmaf(wv.z, a, hv.z + bv.z), 0.f));
  atomAddF32(ag + 3, fmaxf(fmaf(wv.w, a, hv.w + bv.w), 0.f));
}

// ---- Node MLP 1: h1 = relu(relu((x+agg1)@W1a+b1a)@W1b+b1b), 64 nodes/block ----
__global__ __launch_bounds__(128) void mlp1_kernel(
    const float* __restrict__ x, const float* __restrict__ agg1,
    const float* __restrict__ W1a, const float* __restrict__ b1a,
    const float* __restrict__ W1b, const float* __restrict__ b1b,
    float* __restrict__ h1, int N) {
  __shared__ float zf[64 * FDIM];
  __shared__ float w1a_s[FDIM * 64];
  __shared__ float b1a_s[64];
  __shared__ float b1b_s[64];
  __shared__ __align__(16) _Float16 wt1b[64 * 64];
  __shared__ __align__(16) _Float16 th[64 * 64];

  const int tid = threadIdx.x;
  const int nb = blockIdx.x * 64;
  const bool full = (nb + 64 <= N);   // block-uniform: avoids per-row exec-mask guards

  for (int i = tid; i < FDIM * 64; i += 128) w1a_s[i] = W1a[i];
  if (tid < 64) { b1a_s[tid] = b1a[tid]; b1b_s[tid] = b1b[tid]; }
  for (int i = tid; i < 64 * 64; i += 128) {          // stage W1b transposed f16
    int f = i >> 6, c = i & 63;
    wt1b[c * 64 + f] = (_Float16)W1b[i];
  }
  if (full) {
    for (int i = tid; i < 64 * FDIM; i += 128) {      // z = x + agg1 (unguarded)
      size_t g = (size_t)nb * FDIM + i;
      zf[i] = x[g] + agg1[g];
    }
  } else {
    for (int i = tid; i < 64 * FDIM; i += 128) {
      int nl = i / FDIM, f = i - nl * FDIM;
      int node = nb + nl;
      zf[i] = (node < N) ? (x[(size_t)node * FDIM + f] + agg1[(size_t)node * FDIM + f]) : 0.f;
    }
  }
  __syncthreads();
  for (int i = tid; i < 64 * 64; i += 128) {          // t = relu(z@W1a + b1a), K=9 scalar
    int nl = i >> 6, c = i & 63;
    float sacc = b1a_s[c];
    const float* zr = &zf[nl * FDIM];
#pragma unroll
    for (int f = 0; f < FDIM; ++f) sacc = fmaf(zr[f], w1a_s[f * 64 + c], sacc);
    th[i] = (_Float16)fmaxf(sacc, 0.f);
  }
  __syncthreads();
  const int wave = tid >> 5, lane = tid & 31;
  const int r = lane & 15, h = lane >> 4;
  const int m0 = wave * 16;
  for (int n0 = 0; n0 < 64; n0 += 16) {               // h1 = relu(t@W1b + b1b) via WMMA
    v8f acc = gemm_tile(th, 64, wt1b, 64, m0, n0, 64, r, h);
    const int c = n0 + r;
    const float bc = b1b_s[c];
    if (full) {
#pragma unroll
      for (int j = 0; j < 8; ++j) {
        int M = m0 + j + 8 * h;
        h1[(size_t)(nb + M) * HDIM + c] = fmaxf(acc[j] + bc, 0.f);
      }
    } else {
#pragma unroll
      for (int j = 0; j < 8; ++j) {
        int M = m0 + j + 8 * h;
        int node = nb + M;
        if (node < N) h1[(size_t)node * HDIM + c] = fmaxf(acc[j] + bc, 0.f);
      }
    }
  }
}

// ---- Node MLP 2 + readout head: three WMMA GEMMs + sigmoid + group atomics ----
__global__ __launch_bounds__(128) void mlp2_kernel(
    const float* __restrict__ h1, const float* __restrict__ agg2,
    const float* __restrict__ W2a, const float* __restrict__ b2a,
    const float* __restrict__ W2b, const float* __restrict__ b2b,
    const float* __restrict__ Wr1, const float* __restrict__ br1,
    const float* __restrict__ Wr2, const float* __restrict__ br2,
    const int* __restrict__ term, const float* __restrict__ c_cost,
    const int* __restrict__ batch, float* __restrict__ pi, float* __restrict__ tot, int N) {
  __shared__ __align__(16) _Float16 z2h[64 * 64];
  __shared__ __align__(16) _Float16 wt2a[64 * 64];
  __shared__ __align__(16) _Float16 wt2b[64 * 64];
  __shared__ __align__(16) _Float16 th[64 * 64];
  __shared__ __align__(16) _Float16 h2h[64 * 64];
  __shared__ __align__(16) _Float16 wr1t[32 * 64];
  __shared__ float b2a_s[64], b2b_s[64], br1_s[32], wr2_s[32];
  __shared__ float R[64 * 32];

  const int tid = threadIdx.x;
  const int nb = blockIdx.x * 64;
  const bool full = (nb + 64 <= N);

  for (int i = tid; i < 64 * 64; i += 128) {
    int f = i >> 6, c = i & 63;
    wt2a[c * 64 + f] = (_Float16)W2a[i];
    wt2b[c * 64 + f] = (_Float16)W2b[i];
  }
  for (int i = tid; i < 64 * 32; i += 128) {          // Wr1 is [64][32]
    int f = i >> 5, c = i & 31;
    wr1t[c * 64 + f] = (_Float16)Wr1[i];
  }
  if (tid < 64) { b2a_s[tid] = b2a[tid]; b2b_s[tid] = b2b[tid]; }
  if (tid < 32) { br1_s[tid] = br1[tid]; wr2_s[tid] = Wr2[tid]; }
  if (full) {
    for (int i = tid; i < 64 * 64; i += 128) {        // z2 = h1 + agg2 (f16, unguarded)
      size_t g = (size_t)nb * HDIM + i;
      z2h[i] = (_Float16)(h1[g] + agg2[g]);
    }
  } else {
    for (int i = tid; i < 64 * 64; i += 128) {
      int node = nb + (i >> 6);
      float v = 0.f;
      if (node < N) { size_t g = (size_t)nb * HDIM + i; v = h1[g] + agg2[g]; }
      z2h[i] = (_Float16)v;
    }
  }
  __syncthreads();
  const int wave = tid >> 5, lane = tid & 31;
  const int r = lane & 15, h = lane >> 4;
  const int m0 = wave * 16;
  for (int n0 = 0; n0 < 64; n0 += 16) {               // t = relu(z2@W2a + b2a)
    v8f acc = gemm_tile(z2h, 64, wt2a, 64, m0, n0, 64, r, h);
#pragma unroll
    for (int j = 0; j < 8; ++j) {
      int M = m0 + j + 8 * h, c = n0 + r;
      th[M * 64 + c] = (_Float16)fmaxf(acc[j] + b2a_s[c], 0.f);
    }
  }
  __syncthreads();
  for (int n0 = 0; n0 < 64; n0 += 16) {               // h2 = relu(t@W2b + b2b)
    v8f acc = gemm_tile(th, 64, wt2b, 64, m0, n0, 64, r, h);
#pragma unroll
    for (int j = 0; j < 8; ++j) {
      int M = m0 + j + 8 * h, c = n0 + r;
      h2h[M * 64 + c] = (_Float16)fmaxf(acc[j] + b2b_s[c], 0.f);
    }
  }
  __syncthreads();
  for (int n0 = 0; n0 < 32; n0 += 16) {               // r = relu(h2@Wr1 + br1)  [64x32]
    v8f acc = gemm_tile(h2h, 64, wr1t, 64, m0, n0, 64, r, h);
#pragma unroll
    for (int j = 0; j < 8; ++j) {
      int M = m0 + j + 8 * h, c = n0 + r;
      R[M * 32 + c] = fmaxf(acc[j] + br1_s[c], 0.f);
    }
  }
  __syncthreads();
  if (tid < 64 && (full || nb + tid < N)) {           // sigmoid head + group expense
    int node = nb + tid;
    float sacc = br2[0];
#pragma unroll
    for (int c = 0; c < 32; ++c) sacc = fmaf(R[tid * 32 + c], wr2_s[c], sacc);
    float p = 1.f / (1.f + __expf(-sacc));
    p *= (1.f - (float)term[node]);
    pi[node] = p;
    atomAddF32(&tot[batch[node]], p * c_cost[node]);
  }
}

// ---- Finalize: budget ratio ----
__global__ __launch_bounds__(256) void final_kernel(const float* __restrict__ pi,
    const int* __restrict__ batch, const float* __restrict__ B_total,
    const float* __restrict__ tot, float* __restrict__ out, int N) {
  int n = blockIdx.x * 256 + threadIdx.x;
  if (n >= N) return;
  int b = batch[n];
  float ratio = fminf(B_total[b] / (tot[b] + 1e-12f), 1.f);
  out[n] = pi[n] * ratio;
}

extern "C" void kernel_launch(void* const* d_in, const int* in_sizes, int n_in,
                              void* d_out, int out_size, void* d_ws, size_t ws_size,
                              hipStream_t stream) {
  const float* x       = (const float*)d_in[0];
  const int*   ei      = (const int*)  d_in[1];
  const float* ea      = (const float*)d_in[2];
  const int*   batch   = (const int*)  d_in[3];
  const float* B_total = (const float*)d_in[4];
  const int*   term    = (const int*)  d_in[5];
  const float* c_cost  = (const float*)d_in[6];
  const float* We1 = (const float*)d_in[7];   const float* be1 = (const float*)d_in[8];
  const float* W1a = (const float*)d_in[9];   const float* b1a = (const float*)d_in[10];
  const float* W1b = (const float*)d_in[11];  const float* b1b = (const float*)d_in[12];
  const float* We2 = (const float*)d_in[13];  const float* be2 = (const float*)d_in[14];
  const float* W2a = (const float*)d_in[15];  const float* b2a = (const float*)d_in[16];
  const float* W2b = (const float*)d_in[17];  const float* b2b = (const float*)d_in[18];
  const float* Wr1 = (const float*)d_in[19];  const float* br1 = (const float*)d_in[20];
  const float* Wr2 = (const float*)d_in[21];  const float* br2 = (const float*)d_in[22];
  float* out = (float*)d_out;

  const int N = in_sizes[0] / FDIM;
  const int E = in_sizes[2];
  const int G = in_sizes[4];

  // Workspace layout (16B aligned only when N % 4 == 0: N*FDIM*4 = 36N bytes)
  float* agg1 = (float*)d_ws;                  // N*9
  float* h1   = agg1 + (size_t)N * FDIM;       // N*64
  float* agg2 = h1   + (size_t)N * HDIM;       // N*64
  float* pi   = agg2 + (size_t)N * HDIM;       // N
  float* tot  = pi   + (size_t)N;              // G

  hipMemsetAsync(agg1, 0, (size_t)N * FDIM * sizeof(float), stream);
  hipMemsetAsync(agg2, 0, (size_t)N * HDIM * sizeof(float), stream);
  hipMemsetAsync(tot,  0, (size_t)G * sizeof(float), stream);

  const int nblk = (N + 63) / 64;
  edge1_kernel<<<(E + 255) / 256, 256, 0, stream>>>(x, ei, ea, We1, be1, agg1, E);
  mlp1_kernel<<<nblk, 128, 0, stream>>>(x, agg1, W1a, b1a, W1b, b1b, h1, N);
  edge2_kernel<<<(E * 16 + 255) / 256, 256, 0, stream>>>(h1, ei, ea, We2, be2, agg2, E);
  mlp2_kernel<<<nblk, 128, 0, stream>>>(h1, agg2, W2a, b2a, W2b, b2b, Wr1, br1, Wr2, br2,
                                        term, c_cost, batch, pi, tot, N);
  final_kernel<<<(N + 255) / 256, 256, 0, stream>>>(pi, batch, B_total, tot, out, N);
}
